// FusedGromovWasserstein_19550691131581
// MI455X (gfx1250) — matrix operand order repaired and checked by reading, not compiled, not run on hardware
//
#include <hip/hip_runtime.h>
#include <hip/hip_bf16.h>
#include <math.h>

// ---------------------------------------------------------------------------
// Fused Gromov-Wasserstein for MI455X (gfx1250, wave32, WMMA 16x16x32 f16)
// ---------------------------------------------------------------------------

typedef __attribute__((ext_vector_type(16))) _Float16 v16h;
typedef __attribute__((ext_vector_type(8)))  float    v8f;

#define LAMBDA_FGW 0.5f
#define TAU_KL     0.1f
#define N_OUTER    5
#define N_SINK     10
#define SCALE_T    256.0f
#define INV_SCALE_T (1.0f/256.0f)

// ---------------- WMMA fragment helpers (contiguous k = 16h+s for A and B) ---------

__device__ __forceinline__ v8f wmma_f16(v16h a, v16h b, v8f c) {
  return __builtin_amdgcn_wmma_f32_16x16x32_f16(
      /*neg_a=*/false, a, /*neg_b=*/false, b,
      /*c_mod=*/(short)0, c, /*reuse_a=*/false, /*reuse_b=*/false);
}

// A: 16x32 tile, row-major source (lda in elements). Lane L<16 holds row M=L,
// K=0..15; lane L>=16 holds row M=L-16, K=16..31. Contiguous.
__device__ __forceinline__ v16h load_a_f16(const _Float16* A, int lda) {
  int lane = threadIdx.x & 31;
  int m  = lane & 15;
  int kb = (lane >> 4) * 16;
  v16h a;
#pragma unroll
  for (int h = 0; h < 16; ++h) {
    int k = kb + h;
    a[h] = A[m * lda + k];
  }
  return a;
}

__device__ __forceinline__ v16h load_a_f32cvt(const float* A, int lda) {
  int lane = threadIdx.x & 31;
  int m  = lane & 15;
  int kb = (lane >> 4) * 16;
  v16h a;
#pragma unroll
  for (int h = 0; h < 16; ++h) {
    int k = kb + h;
    a[h] = (_Float16)A[m * lda + k];
  }
  return a;
}

// B: 32x16 tile supplied TRANSPOSED (Bt is N x K row-major, ldb in elements).
// Lane L<16: col N=L, K=0..15; lane L>=16: col N=L-16, K=16..31. Contiguous.
__device__ __forceinline__ v16h load_bt_f16(const _Float16* Bt, int ldb) {
  int lane = threadIdx.x & 31;
  int n   = lane & 15;
  int kb2 = (lane >> 4) * 16;
  v16h b;
#pragma unroll
  for (int h = 0; h < 16; ++h) b[h] = Bt[n * ldb + kb2 + h];
  return b;
}

__device__ __forceinline__ v16h load_bt_f32cvt(const float* Bt, int ldb) {
  int lane = threadIdx.x & 31;
  int n   = lane & 15;
  int kb2 = (lane >> 4) * 16;
  v16h b;
#pragma unroll
  for (int h = 0; h < 16; ++h) b[h] = (_Float16)Bt[n * ldb + kb2 + h];
  return b;
}

// ---------------- Kernel 0: transpose+convert W1,W2 -> f16 -----------------

__global__ __launch_bounds__(256) void k_w(const float* W1, const float* W2,
                                           _Float16* W1T, _Float16* W2T) {
  int idx = blockIdx.x * 256 + threadIdx.x;   // 0 .. 524287
  int sel = idx >> 18;
  int e   = idx & 262143;
  int d = e >> 9, j = e & 511;
  if (sel == 0) W1T[j * 512 + d] = (_Float16)W1[d * 512 + j];
  else          W2T[j * 512 + d] = (_Float16)W2[d * 512 + j];
}

// ---------------- Kernel: mu/nu/log_mu/log_nu + rowsum/colsum init ---------

__global__ __launch_bounds__(128) void k_init(const float* mask_q, const float* mask_r,
                                              float* mu, float* nu,
                                              float* lmu, float* lnu,
                                              float* a, float* c) {
  __shared__ float red[128];
  int b = blockIdx.x, t = threadIdx.x;
  float mq = mask_q[b * 128 + t];
  float mr = mask_r[b * 128 + t];
  red[t] = mq; __syncthreads();
  for (int s = 64; s > 0; s >>= 1) { if (t < s) red[t] += red[t + s]; __syncthreads(); }
  float smq = red[0]; __syncthreads();
  red[t] = mr; __syncthreads();
  for (int s = 64; s > 0; s >>= 1) { if (t < s) red[t] += red[t + s]; __syncthreads(); }
  float smr = red[0];
  float muv = mq / (smq + 1e-8f);
  float nuv = mr / (smr + 1e-8f);
  int i = b * 128 + t;
  mu[i] = muv;  nu[i] = nuv;
  lmu[i] = logf(fmaxf(muv, 1e-8f));
  lnu[i] = logf(fmaxf(nuv, 1e-8f));
  a[i] = muv;   c[i] = nuv;   // T0 = mu (x) nu  => rowsum=mu, colsum=nu
}

// ---------------- Kernel: T0 = mu (x) nu, scaled f16 -----------------------

__global__ __launch_bounds__(256) void k_t0(const float* mu, const float* nu,
                                            _Float16* T16) {
  long idx = (long)blockIdx.x * 256 + threadIdx.x;  // B*128*128
  int b = (int)(idx >> 14);
  int k = (int)(idx >> 7) & 127;
  int m = (int)idx & 127;
  T16[idx] = (_Float16)(mu[b * 128 + k] * nu[b * 128 + m] * SCALE_T);
}

// ---------------- Kernel: fused 2-layer MLP (WMMA), 16 rows / block --------
// X[65536,512] (rows 0..32767 = sq, rest = sr);  P16 = relu(X@W1+b1)@W2+b2

__global__ __launch_bounds__(256) void k_mlp(const float* sq, const float* sr,
                                             const _Float16* W1T, const float* b1,
                                             const _Float16* W2T, const float* b2,
                                             _Float16* P16) {
  __shared__ _Float16 lx[16 * 512];
  __shared__ _Float16 lh[16 * 512];
  long row0 = (long)blockIdx.x * 16;
  const float* X = (row0 < 32768) ? (sq + row0 * 512) : (sr + (row0 - 32768) * 512);
  int t = threadIdx.x;
  for (int i = t; i < 16 * 512; i += 256) lx[i] = (_Float16)X[i];
  __syncthreads();
  int wave = t >> 5, lane = t & 31;
  int n = lane & 15, r0 = (lane >> 4) * 8;
  // layer 1: hidden = relu(X @ W1 + b1)  (kept in LDS, f16)
  for (int nt = wave; nt < 32; nt += 8) {
    v8f acc = {};
    for (int kk = 0; kk < 512; kk += 32) {
      v16h a  = load_a_f16(lx + kk, 512);
      v16h bb = load_bt_f16(W1T + (long)nt * 16 * 512 + kk, 512);
      acc = wmma_f16(a, bb, acc);
    }
    float bias = b1[nt * 16 + n];
#pragma unroll
    for (int r = 0; r < 8; ++r) {
      float v = acc[r] + bias;
      lh[(r0 + r) * 512 + nt * 16 + n] = (_Float16)(v > 0.f ? v : 0.f);
    }
  }
  __syncthreads();
  // layer 2: P = hidden @ W2 + b2 -> f16 global
  for (int nt = wave; nt < 32; nt += 8) {
    v8f acc = {};
    for (int kk = 0; kk < 512; kk += 32) {
      v16h a  = load_a_f16(lh + kk, 512);
      v16h bb = load_bt_f16(W2T + (long)nt * 16 * 512 + kk, 512);
      acc = wmma_f16(a, bb, acc);
    }
    float bias = b2[nt * 16 + n];
#pragma unroll
    for (int r = 0; r < 8; ++r)
      P16[(row0 + r0 + r) * 512 + nt * 16 + n] = (_Float16)(acc[r] + bias);
  }
}

// ---------------- Row sum-of-squares (one wave per row) --------------------

__global__ __launch_bounds__(256) void k_sumsq_h(const _Float16* P, float* out) {
  int t = threadIdx.x, lane = t & 31;
  long row = (long)blockIdx.x * 8 + (t >> 5);
  const _Float16* p = P + row * 512;
  float s = 0.f;
  for (int i = lane; i < 512; i += 32) { float v = (float)p[i]; s += v * v; }
  for (int o = 16; o; o >>= 1) s += __shfl_xor(s, o, 32);
  if (lane == 0) out[row] = s;
}

__global__ __launch_bounds__(256) void k_sumsq_f(const float* P, float* out) {
  int t = threadIdx.x, lane = t & 31;
  long row = (long)blockIdx.x * 8 + (t >> 5);
  const float* p = P + row * 512;
  float s = 0.f;
  for (int i = lane; i < 512; i += 32) { float v = p[i]; s += v * v; }
  for (int o = 16; o; o >>= 1) s += __shfl_xor(s, o, 32);
  if (lane == 0) out[row] = s;
}

// ---------------- C = pdist(pq, pr)  (batched WMMA Gram) -------------------

__global__ __launch_bounds__(256) void k_cdist(const _Float16* P16, const float* nP,
                                               float* C) {
  int bx = blockIdx.x;               // 256 batches * 8 row strips
  int b = bx >> 3, k0 = (bx & 7) * 16;
  int t = threadIdx.x, wave = t >> 5, lane = t & 31;
  int m0 = wave * 16;
  const _Float16* A  = P16 + ((long)(b * 128 + k0)) * 512;           // pq rows
  const _Float16* Bt = P16 + ((long)(32768 + b * 128 + m0)) * 512;   // pr rows
  v8f acc = {};
  for (int kk = 0; kk < 512; kk += 32) {
    v16h a  = load_a_f16(A + kk, 512);
    v16h bb = load_bt_f16(Bt + kk, 512);
    acc = wmma_f16(a, bb, acc);
  }
  int n = lane & 15, r0 = (lane >> 4) * 8;
  float nr = nP[32768 + b * 128 + m0 + n];
#pragma unroll
  for (int r = 0; r < 8; ++r) {
    int k = k0 + r0 + r;
    float d2 = nP[b * 128 + k] + nr - 2.f * acc[r];
    C[((long)b * 128 + k) * 128 + m0 + n] = sqrtf(fmaxf(d2, 1e-6f));
  }
}

// ---------------- S = pdist(centroids, centroids) (f32 src, f32+f16 out) ---

__global__ __launch_bounds__(256) void k_gram(const float* Csrc, const float* nC,
                                              float* S, _Float16* S16) {
  int bx = blockIdx.x;
  int b = bx >> 3, k0 = (bx & 7) * 16;
  int t = threadIdx.x, wave = t >> 5, lane = t & 31;
  int m0 = wave * 16;
  const float* A  = Csrc + ((long)(b * 128 + k0)) * 512;
  const float* Bt = Csrc + ((long)(b * 128 + m0)) * 512;
  v8f acc = {};
  for (int kk = 0; kk < 512; kk += 32) {
    v16h a  = load_a_f32cvt(A + kk, 512);
    v16h bb = load_bt_f32cvt(Bt + kk, 512);
    acc = wmma_f16(a, bb, acc);
  }
  int n = lane & 15, r0 = (lane >> 4) * 8;
  float nm = nC[b * 128 + m0 + n];
#pragma unroll
  for (int r = 0; r < 8; ++r) {
    int k = k0 + r0 + r;
    float d2 = nC[b * 128 + k] + nm - 2.f * acc[r];
    float s = sqrtf(fmaxf(d2, 1e-6f));
    long idx = ((long)b * 128 + k) * 128 + m0 + n;
    S[idx] = s;
    S16[idx] = (_Float16)s;
  }
}

// ---------------- t1 = Sq^2 @ rowsum(T),  t2 = colsum(T) @ Sr^2 ------------

__global__ __launch_bounds__(256) void k_t12(const float* Sq, const float* Sr,
                                             const float* a, const float* c,
                                             float* t1, float* t2) {
  __shared__ float la[128], lc[128];
  int b = blockIdx.x, t = threadIdx.x;
  if (t < 128) la[t] = a[b * 128 + t];
  else         lc[t - 128] = c[b * 128 + (t - 128)];
  __syncthreads();
  if (t < 128) {
    const float* row = Sq + ((long)b * 128 + t) * 128;
    float s = 0.f;
    for (int l = 0; l < 128; ++l) { float v = row[l]; s += v * v * la[l]; }
    t1[b * 128 + t] = s;
  } else {
    int n = t - 128;
    const float* Srb = Sr + (long)b * 128 * 128;
    float s = 0.f;
    for (int m = 0; m < 128; ++m) { float v = Srb[m * 128 + n]; s += v * v * lc[m]; }
    t2[b * 128 + n] = s;
  }
}

// ---------------- TSr = T @ Sr  (Sr symmetric), stored TRANSPOSED f16 ------

__global__ __launch_bounds__(256) void k_tsr(const _Float16* T16, const _Float16* Sr16,
                                             _Float16* TSrT16) {
  int bx = blockIdx.x;
  int b = bx >> 3, k0 = (bx & 7) * 16;
  int t = threadIdx.x, wave = t >> 5, lane = t & 31;
  int n0 = wave * 16;
  const _Float16* A  = T16  + ((long)b * 128 + k0) * 128;
  const _Float16* Bt = Sr16 + ((long)b * 128 + n0) * 128;   // symmetric
  v8f acc = {};
  for (int kk = 0; kk < 128; kk += 32) {
    v16h a  = load_a_f16(A + kk, 128);
    v16h bb = load_bt_f16(Bt + kk, 128);
    acc = wmma_f16(a, bb, acc);
  }
  int n = lane & 15, r0 = (lane >> 4) * 8;
#pragma unroll
  for (int r = 0; r < 8; ++r)
    TSrT16[((long)b * 128 + n0 + n) * 128 + k0 + r0 + r] = (_Float16)acc[r];
}

// ---------------- C_fgw and log_K:  G = Sq @ TSr (Sq symmetric) ------------

__global__ __launch_bounds__(256) void k_cfgw(const _Float16* Sq16, const _Float16* TSrT16,
                                              const float* Cmat, const float* t1,
                                              const float* t2, const float* log_eps,
                                              float* logK) {
  int bx = blockIdx.x;
  int b = bx >> 3, k0 = (bx & 7) * 16;
  int t = threadIdx.x, wave = t >> 5, lane = t & 31;
  int m0 = wave * 16;
  const _Float16* A  = Sq16   + ((long)b * 128 + k0) * 128;
  const _Float16* Bt = TSrT16 + ((long)b * 128 + m0) * 128;
  v8f acc = {};
  for (int kk = 0; kk < 128; kk += 32) {
    v16h a  = load_a_f16(A + kk, 128);
    v16h bb = load_bt_f16(Bt + kk, 128);
    acc = wmma_f16(a, bb, acc);
  }
  float eps = fminf(fmaxf(expf(log_eps[0]), 0.01f), 0.5f);
  int n = lane & 15, r0 = (lane >> 4) * 8;
  float t2v = t2[b * 128 + m0 + n];
#pragma unroll
  for (int r = 0; r < 8; ++r) {
    int k = k0 + r0 + r;
    float G   = acc[r] * INV_SCALE_T;
    float Lgw = t1[b * 128 + k] + t2v - 2.f * G;
    long idx = ((long)b * 128 + k) * 128 + m0 + n;
    float cf = (1.0f - LAMBDA_FGW) * Cmat[idx] + LAMBDA_FGW * Lgw;
    logK[idx] = -cf / eps;
  }
}

// ---------------- Sinkhorn: whole 128x128 log_K in LDS (CDNA5 320KB) -------

#define SK_PITCH 132   // coprime-ish with 64 banks -> conflict-free columns

__global__ __launch_bounds__(256) void k_sinkhorn(const float* logK,
                                                  const float* lmu_g, const float* lnu_g,
                                                  const float* log_eps,
                                                  float* T_out, _Float16* T16,
                                                  float* a_out, float* c_out,
                                                  float* cost_out) {
  __shared__ float lK[128 * SK_PITCH];
  __shared__ float lu[128], lv[128], lmu[128], lnu[128];
  __shared__ float lacc[128], lcc[128];
  __shared__ float red[256];
  int b = blockIdx.x, t = threadIdx.x;
  const float* src = logK + (long)b * 16384;
  for (int i = t; i < 16384; i += 256)
    lK[(i >> 7) * SK_PITCH + (i & 127)] = src[i];
  if (t < 128) {
    lmu[t] = lmu_g[b * 128 + t];
    lnu[t] = lnu_g[b * 128 + t];
    lu[t] = 0.f; lv[t] = 0.f; lacc[t] = 0.f; lcc[t] = 0.f;
  }
  __syncthreads();
  float eps = fminf(fmaxf(expf(log_eps[0]), 0.01f), 0.5f);
  float rho = TAU_KL / (TAU_KL + eps);
  for (int it = 0; it < N_SINK; ++it) {
    if (t < 128) {                       // log_u over rows
      const float* row = lK + t * SK_PITCH;
      float mx = -1e30f;
      for (int m = 0; m < 128; ++m) mx = fmaxf(mx, row[m] + lv[m]);
      float s = 0.f;
      for (int m = 0; m < 128; ++m) s += expf(row[m] + lv[m] - mx);
      lu[t] = rho * (lmu[t] - (mx + logf(s)));
    }
    __syncthreads();
    if (t < 128) {                       // log_v over cols
      float mx = -1e30f;
      for (int k = 0; k < 128; ++k) mx = fmaxf(mx, lK[k * SK_PITCH + t] + lu[k]);
      float s = 0.f;
      for (int k = 0; k < 128; ++k) s += expf(lK[k * SK_PITCH + t] + lu[k] - mx);
      lv[t] = rho * (lnu[t] - (mx + logf(s)));
    }
    __syncthreads();
  }
  // T = exp(lu + logK + lv); accumulate rowsums, colsums, cost
  float cpart = 0.f;
  for (int i = t; i < 16384; i += 256) {
    int k = i >> 7, m = i & 127;
    float lkv = lK[k * SK_PITCH + m];
    float tv  = expf(lu[k] + lkv + lv[m]);
    T_out[(long)b * 16384 + i] = tv;
    T16[(long)b * 16384 + i]   = (_Float16)(tv * SCALE_T);
    atomicAdd(&lacc[k], tv);
    atomicAdd(&lcc[m], tv);
    cpart += tv * (-eps * lkv);          // C_fgw = -eps * log_K
  }
  red[t] = cpart; __syncthreads();
  for (int s = 128; s > 0; s >>= 1) { if (t < s) red[t] += red[t + s]; __syncthreads(); }
  if (t == 0) cost_out[b] = red[0];
  if (t < 128) { a_out[b * 128 + t] = lacc[t]; c_out[b * 128 + t] = lcc[t]; }
}

// ---------------- similarity = sigmoid(-cost) ------------------------------

__global__ __launch_bounds__(256) void k_final(const float* cost, float* sim_out,
                                               float* cost_out) {
  int t = threadIdx.x;   // 256 batches
  float cst = cost[t];
  sim_out[t]  = 1.f / (1.f + expf(cst));
  cost_out[t] = cst;
}

// ---------------------------------------------------------------------------

extern "C" void kernel_launch(void* const* d_in, const int* in_sizes, int n_in,
                              void* d_out, int out_size, void* d_ws, size_t ws_size,
                              hipStream_t stream) {
  const float* sq      = (const float*)d_in[0];
  const float* sr      = (const float*)d_in[1];
  const float* mask_q  = (const float*)d_in[2];
  const float* mask_r  = (const float*)d_in[3];
  const float* cq      = (const float*)d_in[4];
  const float* cr      = (const float*)d_in[5];
  const float* W1      = (const float*)d_in[6];
  const float* b1      = (const float*)d_in[7];
  const float* W2      = (const float*)d_in[8];
  const float* b2      = (const float*)d_in[9];
  const float* log_eps = (const float*)d_in[10];

  float* out = (float*)d_out;
  const long NT = 256L * 128 * 128;           // 4,194,304
  float* out_sim  = out;                      // [256]
  float* out_T    = out + 256;                // [B,K,M]
  float* out_C    = out + 256 + NT;           // [B,K,M]
  float* out_cost = out + 256 + 2 * NT;       // [256]

  char* ws = (char*)d_ws;
  _Float16* W1T   = (_Float16*)(ws + 0);              //  512 KB
  _Float16* W2T   = (_Float16*)(ws + (1L << 19));     //  512 KB
  float*    nP    = (float*)   (ws + 1048576L);       //  256 KB (65536 rows)
  float*    ncq   = (float*)   (ws + 1310720L);       //  128 KB
  float*    ncr   = (float*)   (ws + 1441792L);       //  128 KB
  float*    mu    = (float*)   (ws + 1572864L);
  float*    nu    = (float*)   (ws + 1703936L);
  float*    lmu   = (float*)   (ws + 1835008L);
  float*    lnu   = (float*)   (ws + 1966080L);
  float*    asum  = (float*)   (ws + 2097152L);
  float*    csum  = (float*)   (ws + 2228224L);
  float*    t1    = (float*)   (ws + 2359296L);
  float*    t2    = (float*)   (ws + 2490368L);
  float*    costw = (float*)   (ws + 2621440L);
  float*    Sq    = (float*)   (ws + 4194304L);       // 16 MB
  float*    Sr    = (float*)   (ws + 20971520L);      // 16 MB
  _Float16* Sq16  = (_Float16*)(ws + 37748736L);      //  8 MB
  _Float16* Sr16  = (_Float16*)(ws + 46137344L);      //  8 MB
  _Float16* T16   = (_Float16*)(ws + 54525952L);      //  8 MB
  _Float16* TSrT  = (_Float16*)(ws + 62914560L);      //  8 MB
  float*    logK  = (float*)   (ws + 71303168L);      // 16 MB
  _Float16* P16   = (_Float16*)(ws + 88080384L);      // 64 MB (65536x512)

  dim3 blk256(256), blk128(128);

  // prologue
  k_w<<<2048, blk256, 0, stream>>>(W1, W2, W1T, W2T);
  k_init<<<256, blk128, 0, stream>>>(mask_q, mask_r, mu, nu, lmu, lnu, asum, csum);
  k_t0<<<16384, blk256, 0, stream>>>(mu, nu, T16);
  k_mlp<<<4096, blk256, 0, stream>>>(sq, sr, W1T, b1, W2T, b2, P16);
  k_sumsq_h<<<8192, blk256, 0, stream>>>(P16, nP);
  k_sumsq_f<<<4096, blk256, 0, stream>>>(cq, ncq);
  k_sumsq_f<<<4096, blk256, 0, stream>>>(cr, ncr);
  k_cdist<<<2048, blk256, 0, stream>>>(P16, nP, out_C);
  k_gram<<<2048, blk256, 0, stream>>>(cq, ncq, Sq, Sq16);
  k_gram<<<2048, blk256, 0, stream>>>(cr, ncr, Sr, Sr16);

  // FGW outer loop
  for (int it = 0; it < N_OUTER; ++it) {
    k_t12<<<256, blk256, 0, stream>>>(Sq, Sr, asum, csum, t1, t2);
    k_tsr<<<2048, blk256, 0, stream>>>(T16, Sr16, TSrT);
    k_cfgw<<<2048, blk256, 0, stream>>>(Sq16, TSrT, out_C, t1, t2, log_eps, logK);
    k_sinkhorn<<<256, blk256, 0, stream>>>(logK, lmu, lnu, log_eps,
                                           out_T, T16, asum, csum, costw);
  }

  k_final<<<1, blk256, 0, stream>>>(costw, out_sim, out_cost);

  (void)in_sizes; (void)n_in; (void)out_size; (void)ws_size;
}
